// BahdanauAttentionAudio_16612933501325
// MI455X (gfx1250) — matrix operand order repaired and checked by reading, not compiled, not run on hardware
//
#include <hip/hip_runtime.h>
#include <math.h>

// Problem constants (from reference): B=20, S=198, H=4096, K=15, TOPK=100
#define BB 20
#define SS 198
#define HH 4096
#define MM (BB * SS)   // 3960 rows of the big GEMM
#define TOPK_N 100

typedef __attribute__((ext_vector_type(16))) __bf16 v16bf;
typedef __attribute__((ext_vector_type(8)))  __bf16 v8bf;
typedef __attribute__((ext_vector_type(4)))  __bf16 v4bf;
typedef __attribute__((ext_vector_type(8)))  float  v8f;

// ---------------------------------------------------------------------------
// Fragment loaders from pre-split bf16 arrays (natural [row][k] layout).
// A (16-bit 16x32 ISA layout): lane<16 holds K[0..7]|K[16..23], lane>=16 K[8..15]|K[24..31]
//   -> base already offset by half*8; chunks at +0 and +16 elements (16B each, 16B aligned)
// B (same k order as A): lane<16 -> K[0..7]|K[16..23], lane>=16 -> K[8..15]|K[24..31]
//   -> base already offset by half*8; chunks at +0 and +16 elements
// ---------------------------------------------------------------------------
__device__ __forceinline__ v16bf ldfragA(const __bf16* __restrict__ p) {
    union { v16bf v; v8bf h[2]; } u;
    u.h[0] = *(const v8bf*)(p);
    u.h[1] = *(const v8bf*)(p + 16);
    return u.v;
}
__device__ __forceinline__ v16bf ldfragB(const __bf16* __restrict__ p) {
    union { v16bf v; v8bf h[2]; } u;
    u.h[0] = *(const v8bf*)(p);
    u.h[1] = *(const v8bf*)(p + 16);
    return u.v;
}

// Convert 8 contiguous f32 into bf16 hi + residual lo fragments (fallback path).
__device__ __forceinline__ void cvt8(const float* __restrict__ p, v16bf& hi, v16bf& lo, int base) {
    float4 x0 = *(const float4*)(p);
    float4 x1 = *(const float4*)(p + 4);
    float v[8] = {x0.x, x0.y, x0.z, x0.w, x1.x, x1.y, x1.z, x1.w};
#pragma unroll
    for (int i = 0; i < 8; ++i) {
        float f   = v[i];
        __bf16 h  = (__bf16)f;
        hi[base + i] = h;
        lo[base + i] = (__bf16)(f - (float)h);
    }
}

// ---------------------------------------------------------------------------
// k0: one-shot split of an f32 array into bf16 hi + residual lo (4 elems/thread)
// ---------------------------------------------------------------------------
__global__ __launch_bounds__(256)
void split_bf16_kernel(const float* __restrict__ src, __bf16* __restrict__ hi,
                       __bf16* __restrict__ lo, int n4) {
    int i = blockIdx.x * 256 + threadIdx.x;
    if (i >= n4) return;
    float4 x = ((const float4*)src)[i];
    float v[4] = {x.x, x.y, x.z, x.w};
    v4bf hv, lv;
#pragma unroll
    for (int j = 0; j < 4; ++j) {
        __bf16 h = (__bf16)v[j];
        hv[j] = h;
        lv[j] = (__bf16)(v[j] - (float)h);
    }
    ((v4bf*)hi)[i] = hv;   // 8-byte stores
    ((v4bf*)lo)[i] = lv;
}

// ---------------------------------------------------------------------------
// k1: degenerate conv (center tap only) + zero the score accumulator
// ---------------------------------------------------------------------------
__global__ void convo_init_kernel(const float* __restrict__ prev_att,
                                  const float* __restrict__ conv_w,
                                  float* __restrict__ convo,
                                  float* __restrict__ score_ws) {
    int i = blockIdx.x * blockDim.x + threadIdx.x;
    if (i >= MM) return;
    score_ws[i] = 0.f;
    int b = i / SS, s = i % SS;
    const float* pa = prev_att + b * SS;
    float acc = 0.f;
    for (int j = 0; j < SS; ++j)
        acc += pa[j] * conv_w[(s * SS + j) * 31 + 15];
    convo[i] = acc;
}

// ---------------------------------------------------------------------------
// k2: qproj[b,h] = q[b,:] . W2_w[h,:] + W2_b[h]   (one wave per h, lanes over k)
// ---------------------------------------------------------------------------
__global__ __launch_bounds__(256)
void qproj_kernel(const float* __restrict__ q, const float* __restrict__ W2,
                  const float* __restrict__ W2b, float* __restrict__ qproj) {
    int gw   = (blockIdx.x * 256 + threadIdx.x) >> 5;
    int lane = threadIdx.x & 31;
    if (gw >= HH) return;
    const float* wrow = W2 + (size_t)gw * HH;
    float accb[BB];
#pragma unroll
    for (int b = 0; b < BB; ++b) accb[b] = 0.f;
    for (int k = lane; k < HH; k += 32) {
        float w = wrow[k];
#pragma unroll
        for (int b = 0; b < BB; ++b) accb[b] += q[b * HH + k] * w;
    }
    float bias = W2b[gw];
#pragma unroll
    for (int b = 0; b < BB; ++b) {
        float v = accb[b];
#pragma unroll
        for (int off = 1; off < 32; off <<= 1) v += __shfl_xor(v, off, 32);
        if (lane == 0) qproj[b * HH + gw] = v + bias;
    }
}

// ---------------------------------------------------------------------------
// Shared GEMM epilogue: bias + qproj + loc term, tanh, dot with V_w, reduce
// across the 16 N-lanes, atomically accumulate per-row score partials.
// C layout: VGPR r -> M = r + 8*half, N = lr.
// ---------------------------------------------------------------------------
__device__ __forceinline__ void score_epilogue(
        v8f acc[2][2], int mbase, int half, int lr, const int hcol[2],
        const float* __restrict__ W1b,    const float* __restrict__ Vw,
        const float* __restrict__ locproj,const float* __restrict__ convo,
        const float* __restrict__ qproj,  float* __restrict__ score_ws) {
    float w1b[2], lp[2], vw[2];
#pragma unroll
    for (int nt = 0; nt < 2; ++nt) {
        int h = hcol[nt];
        w1b[nt] = W1b[h]; lp[nt] = locproj[h]; vw[nt] = Vw[h];
    }
#pragma unroll
    for (int mt = 0; mt < 2; ++mt) {
#pragma unroll
        for (int r = 0; r < 8; ++r) {
            int  m     = mbase + mt * 16 + r + 8 * half;
            bool valid = (m < MM);
            int  mm    = valid ? m : (MM - 1);
            int  bidx  = mm / SS;
            float cv   = convo[mm];
            float contrib = 0.f;
#pragma unroll
            for (int nt = 0; nt < 2; ++nt) {
                float x = acc[mt][nt][r] + w1b[nt]
                        + qproj[bidx * HH + hcol[nt]] + cv * lp[nt];
                contrib += tanhf(x) * vw[nt];
            }
#pragma unroll
            for (int off = 1; off < 16; off <<= 1)
                contrib += __shfl_xor(contrib, off, 32);
            if (lr == 0 && valid) atomicAdd(score_ws + m, contrib);
        }
    }
}

// ---------------------------------------------------------------------------
// k3 (fast path): GEMM on pre-split bf16 hi/lo. Hot loop = pure b128 loads +
// 12 WMMAs per K-step; zero conversion VALU.
// Wave tile 32x32 (2x2 of 16x16), WG = 8 waves -> 64M x 128N, full K sweep.
// ---------------------------------------------------------------------------
__global__ __launch_bounds__(256)
void score_gemm_presplit(const __bf16* __restrict__ vhi, const __bf16* __restrict__ vlo,
                         const __bf16* __restrict__ w1hi,const __bf16* __restrict__ w1lo,
                         const float* __restrict__ W1b,  const float* __restrict__ Vw,
                         const float* __restrict__ locproj,const float* __restrict__ convo,
                         const float* __restrict__ qproj,  float* __restrict__ score_ws) {
    const int lane = threadIdx.x & 31;
    const int wave = threadIdx.x >> 5;
    const int half = lane >> 4;
    const int lr   = lane & 15;
    const int wm   = wave & 1;
    const int wn   = wave >> 1;
    const int mbase = blockIdx.x * 64 + wm * 32;
    const int hbase = blockIdx.y * 128 + wn * 32;

    v8f acc[2][2] = {};

    int mrow[2], hcol[2];
#pragma unroll
    for (int mt = 0; mt < 2; ++mt) {
        int m = mbase + mt * 16 + lr;
        mrow[mt] = (m < MM) ? m : (MM - 1);
    }
#pragma unroll
    for (int nt = 0; nt < 2; ++nt) hcol[nt] = hbase + nt * 16 + lr;

    const __bf16* ah[2], *al[2], *bh[2], *bl[2];
#pragma unroll
    for (int mt = 0; mt < 2; ++mt) {
        size_t off = (size_t)mrow[mt] * HH + half * 8;
        ah[mt] = vhi + off; al[mt] = vlo + off;
    }
#pragma unroll
    for (int nt = 0; nt < 2; ++nt) {
        size_t off = (size_t)hcol[nt] * HH + half * 8;
        bh[nt] = w1hi + off; bl[nt] = w1lo + off;
    }

#pragma unroll 2
    for (int k0 = 0; k0 < HH; k0 += 32) {
        v16bf ahi[2], alo[2], bhi[2], blo[2];
#pragma unroll
        for (int mt = 0; mt < 2; ++mt) {
            ahi[mt] = ldfragA(ah[mt] + k0);
            alo[mt] = ldfragA(al[mt] + k0);
        }
#pragma unroll
        for (int nt = 0; nt < 2; ++nt) {
            bhi[nt] = ldfragB(bh[nt] + k0);
            blo[nt] = ldfragB(bl[nt] + k0);
        }
        __builtin_prefetch(ah[0] + k0 + 64, 0, 1);
        __builtin_prefetch(ah[1] + k0 + 64, 0, 1);
        __builtin_prefetch(bh[0] + k0 + 64, 0, 1);
        __builtin_prefetch(bh[1] + k0 + 64, 0, 1);

        // three passes: same-accumulator WMMAs are 4 apart, not back-to-back
#pragma unroll
        for (int mt = 0; mt < 2; ++mt)
#pragma unroll
            for (int nt = 0; nt < 2; ++nt)
                acc[mt][nt] = __builtin_amdgcn_wmma_f32_16x16x32_bf16(
                    false, ahi[mt], false, bhi[nt], (short)0, acc[mt][nt], false, false);
#pragma unroll
        for (int mt = 0; mt < 2; ++mt)
#pragma unroll
            for (int nt = 0; nt < 2; ++nt)
                acc[mt][nt] = __builtin_amdgcn_wmma_f32_16x16x32_bf16(
                    false, ahi[mt], false, blo[nt], (short)0, acc[mt][nt], false, false);
#pragma unroll
        for (int mt = 0; mt < 2; ++mt)
#pragma unroll
            for (int nt = 0; nt < 2; ++nt)
                acc[mt][nt] = __builtin_amdgcn_wmma_f32_16x16x32_bf16(
                    false, alo[mt], false, bhi[nt], (short)0, acc[mt][nt], false, false);
    }

    score_epilogue(acc, mbase, half, lr, hcol, W1b, Vw, locproj, convo, qproj, score_ws);
}

// ---------------------------------------------------------------------------
// k3 (fallback): convert-in-loop version, used only if ws is too small for
// the pre-split arrays.
// ---------------------------------------------------------------------------
__global__ __launch_bounds__(256)
void score_gemm_inline(const float* __restrict__ values, const float* __restrict__ W1,
                       const float* __restrict__ W1b,    const float* __restrict__ Vw,
                       const float* __restrict__ locproj,const float* __restrict__ convo,
                       const float* __restrict__ qproj,  float* __restrict__ score_ws) {
    const int lane = threadIdx.x & 31;
    const int wave = threadIdx.x >> 5;
    const int half = lane >> 4;
    const int lr   = lane & 15;
    const int wm   = wave & 1;
    const int wn   = wave >> 1;
    const int mbase = blockIdx.x * 64 + wm * 32;
    const int hbase = blockIdx.y * 128 + wn * 32;

    v8f acc[2][2] = {};

    int mrow[2], hcol[2];
#pragma unroll
    for (int mt = 0; mt < 2; ++mt) {
        int m = mbase + mt * 16 + lr;
        mrow[mt] = (m < MM) ? m : (MM - 1);
    }
#pragma unroll
    for (int nt = 0; nt < 2; ++nt) hcol[nt] = hbase + nt * 16 + lr;

    const float* aptr0 = values + (size_t)mrow[0] * HH + half * 8;
    const float* aptr1 = values + (size_t)mrow[1] * HH + half * 8;
    const float* bptr0 = W1 + (size_t)hcol[0] * HH + half * 8;
    const float* bptr1 = W1 + (size_t)hcol[1] * HH + half * 8;

    for (int k0 = 0; k0 < HH; k0 += 32) {
        v16bf ahi[2], alo[2], bhi[2], blo[2];
        cvt8(aptr0 + k0,      ahi[0], alo[0], 0);
        cvt8(aptr0 + k0 + 16, ahi[0], alo[0], 8);
        cvt8(aptr1 + k0,      ahi[1], alo[1], 0);
        cvt8(aptr1 + k0 + 16, ahi[1], alo[1], 8);
        cvt8(bptr0 + k0,      bhi[0], blo[0], 0);
        cvt8(bptr0 + k0 + 16, bhi[0], blo[0], 8);
        cvt8(bptr1 + k0,      bhi[1], blo[1], 0);
        cvt8(bptr1 + k0 + 16, bhi[1], blo[1], 8);

#pragma unroll
        for (int mt = 0; mt < 2; ++mt)
#pragma unroll
            for (int nt = 0; nt < 2; ++nt) {
                acc[mt][nt] = __builtin_amdgcn_wmma_f32_16x16x32_bf16(
                    false, ahi[mt], false, bhi[nt], (short)0, acc[mt][nt], false, false);
                acc[mt][nt] = __builtin_amdgcn_wmma_f32_16x16x32_bf16(
                    false, ahi[mt], false, blo[nt], (short)0, acc[mt][nt], false, false);
                acc[mt][nt] = __builtin_amdgcn_wmma_f32_16x16x32_bf16(
                    false, alo[mt], false, bhi[nt], (short)0, acc[mt][nt], false, false);
            }
    }

    score_epilogue(acc, mbase, half, lr, hcol, W1b, Vw, locproj, convo, qproj, score_ws);
}

// ---------------------------------------------------------------------------
// k4: per-batch top-100 by rank counting, masked score + sigmoid
// ---------------------------------------------------------------------------
__global__ __launch_bounds__(256)
void topk_kernel(const float* __restrict__ score_ws, const float* __restrict__ Vb,
                 float* __restrict__ score_out, float* __restrict__ sig_ws) {
    __shared__ float sv[SS];
    int b = blockIdx.x;
    int t = threadIdx.x;
    if (t < SS) sv[t] = score_ws[b * SS + t] + Vb[0];
    __syncthreads();
    if (t < SS) {
        float v = sv[t];
        int rank = 0;
        for (int j = 0; j < SS; ++j) {
            float u = sv[j];
            rank += (u > v) || (u == v && j < t);
        }
        float ov = (rank < TOPK_N) ? v : 0.f;
        score_out[b * SS + t] = ov;
        sig_ws[b * SS + t]    = 1.f / (1.f + __expf(-ov));
    }
}

// ---------------------------------------------------------------------------
// k5: attention_weights[b,s] = sig[b,s] / sum_b' sig[b',s]
// ---------------------------------------------------------------------------
__global__ void aw_kernel(const float* __restrict__ sig, float* __restrict__ aw_out) {
    int i = blockIdx.x * blockDim.x + threadIdx.x;
    if (i >= MM) return;
    int s = i % SS;
    float d = 0.f;
#pragma unroll
    for (int b = 0; b < BB; ++b) d += sig[b * SS + s];
    aw_out[i] = sig[i] / d;
}

// ---------------------------------------------------------------------------
// k6: context[b,h] = sum_s aw[b,s] * values[b,s,h]
// ---------------------------------------------------------------------------
__global__ __launch_bounds__(256)
void ctx_kernel(const float* __restrict__ values, const float* __restrict__ aw,
                float* __restrict__ ctx) {
    int i = blockIdx.x * blockDim.x + threadIdx.x;
    if (i >= BB * HH) return;
    int b = i >> 12;
    int h = i & (HH - 1);
    const float* vp = values + (size_t)b * SS * HH + h;
    const float* ap = aw + b * SS;
    float acc = 0.f;
    for (int s = 0; s < SS; ++s) acc += ap[s] * vp[(size_t)s * HH];
    ctx[i] = acc;
}

// ---------------------------------------------------------------------------
// launch
// ---------------------------------------------------------------------------
extern "C" void kernel_launch(void* const* d_in, const int* in_sizes, int n_in,
                              void* d_out, int out_size, void* d_ws, size_t ws_size,
                              hipStream_t stream) {
    const float* query    = (const float*)d_in[0];
    const float* values   = (const float*)d_in[1];
    const float* prev_att = (const float*)d_in[2];
    const float* W1_w     = (const float*)d_in[3];
    const float* W1_b     = (const float*)d_in[4];
    const float* W2_w     = (const float*)d_in[5];
    const float* W2_b     = (const float*)d_in[6];
    const float* V_w      = (const float*)d_in[7];
    const float* V_b      = (const float*)d_in[8];
    const float* conv_w   = (const float*)d_in[9];
    const float* locproj  = (const float*)d_in[10];

    float* out       = (float*)d_out;
    float* ctx_out   = out;                  // B*H
    float* aw_out    = out + BB * HH;        // B*S
    float* score_out = aw_out + MM;          // B*S

    const size_t nV = (size_t)MM * HH;       // values elements
    const size_t nW = (size_t)HH * HH;       // W1 elements
    const size_t small_f32 = (size_t)3 * MM + (size_t)BB * HH;   // score+convo+sig+qproj
    const size_t need_big  = (2 * nV + 2 * nW) * sizeof(__bf16) + small_f32 * sizeof(float);

    dim3 grid(62, 32);   // 62 M-tiles of 64 (3968>=3960), 32 N-tiles of 128 (=4096)

    if (ws_size >= need_big) {
        // fast path: pre-split bf16 hi/lo in workspace (all offsets 16B aligned)
        __bf16* vhi  = (__bf16*)d_ws;
        __bf16* vlo  = vhi + nV;
        __bf16* w1hi = vlo + nV;
        __bf16* w1lo = w1hi + nW;
        float* score_ws = (float*)(w1lo + nW);
        float* convo_ws = score_ws + MM;
        float* qproj_ws = convo_ws + MM;
        float* sig_ws   = qproj_ws + BB * HH;

        split_bf16_kernel<<<(int)((nV / 4 + 255) / 256), 256, 0, stream>>>(values, vhi, vlo, (int)(nV / 4));
        split_bf16_kernel<<<(int)((nW / 4 + 255) / 256), 256, 0, stream>>>(W1_w, w1hi, w1lo, (int)(nW / 4));
        convo_init_kernel<<<(MM + 255) / 256, 256, 0, stream>>>(prev_att, conv_w, convo_ws, score_ws);
        qproj_kernel<<<(HH * 32 + 255) / 256, 256, 0, stream>>>(query, W2_w, W2_b, qproj_ws);
        score_gemm_presplit<<<grid, 256, 0, stream>>>(vhi, vlo, w1hi, w1lo, W1_b, V_w, locproj,
                                                      convo_ws, qproj_ws, score_ws);
        topk_kernel<<<BB, 256, 0, stream>>>(score_ws, V_b, score_out, sig_ws);
        aw_kernel<<<(MM + 255) / 256, 256, 0, stream>>>(sig_ws, aw_out);
        ctx_kernel<<<(BB * HH + 255) / 256, 256, 0, stream>>>(values, aw_out, ctx_out);
    } else {
        // fallback: convert inside the GEMM loop (small workspace only)
        float* score_ws = (float*)d_ws;
        float* convo_ws = score_ws + MM;
        float* qproj_ws = convo_ws + MM;
        float* sig_ws   = qproj_ws + BB * HH;

        convo_init_kernel<<<(MM + 255) / 256, 256, 0, stream>>>(prev_att, conv_w, convo_ws, score_ws);
        qproj_kernel<<<(HH * 32 + 255) / 256, 256, 0, stream>>>(query, W2_w, W2_b, qproj_ws);
        score_gemm_inline<<<grid, 256, 0, stream>>>(values, W1_w, W1_b, V_w, locproj,
                                                    convo_ws, qproj_ws, score_ws);
        topk_kernel<<<BB, 256, 0, stream>>>(score_ws, V_b, score_out, sig_ws);
        aw_kernel<<<(MM + 255) / 256, 256, 0, stream>>>(sig_ws, aw_out);
        ctx_kernel<<<(BB * HH + 255) / 256, 256, 0, stream>>>(values, aw_out, ctx_out);
    }
}
